// EfficientTemporalBlock_48567490183260
// MI455X (gfx1250) — matrix operand order repaired and checked by reading, not compiled, not run on hardware
//
#include <hip/hip_runtime.h>
#include <math.h>

// ---------------------------------------------------------------------------
// EfficientTemporalBlock for MI455X (gfx1250, wave32, WMMA, TDM)
// bf16x3-split WMMA (f32 accumulate); x hi/lo split precomputed; register-
// blocked 16x64 wave tiles; TDM double-buffered prefetch in the LSTM scan.
// ---------------------------------------------------------------------------

typedef __attribute__((ext_vector_type(16))) __bf16 v16bf;
typedef __attribute__((ext_vector_type(8)))  __bf16 v8bf;
typedef __attribute__((ext_vector_type(8)))  float  v8f;
typedef __attribute__((ext_vector_type(4)))  unsigned int u32x4;
typedef __attribute__((ext_vector_type(8)))  int i32x8;
typedef __attribute__((ext_vector_type(4)))  int i32x4;

#define Bsz 32
#define Tsz 512
#define Dsz 512
#define Hsz 256
#define G4  1024   // 4*H
#define KC  384    // conv K = 128 channels * 3 taps

#if defined(__HIP_DEVICE_COMPILE__) && __has_builtin(__builtin_amdgcn_tensor_load_to_lds) && __has_builtin(__builtin_amdgcn_s_wait_tensorcnt)
#define USE_TDM 1
#else
#define USE_TDM 0
#endif

static __device__ __forceinline__ v16bf cat8(v8bf a, v8bf b) {
  v16bf r;
#pragma unroll
  for (int j = 0; j < 8; ++j) { r[j] = a[j]; r[j + 8] = b[j]; }
  return r;
}

static __device__ __forceinline__ v8f wmma_bf16(v16bf a, v16bf b, v8f c) {
  return __builtin_amdgcn_wmma_f32_16x16x32_bf16(
      false, a, false, b, (short)0, c, false, false);
}

// A-fragment (16-bit A 16x32 layout): two contiguous 8-elem runs at +0 / +16
static __device__ __forceinline__ v16bf afrag(const __bf16* p) {
  return cat8(*(const v8bf*)p, *(const v8bf*)(p + 16));
}
// B-fragment (16-bit B 32x16 layout): two contiguous 8-elem runs at +0 / +16 (base khalf*8)
static __device__ __forceinline__ v16bf bfrag(const __bf16* p) {
  return cat8(*(const v8bf*)p, *(const v8bf*)(p + 16));
}

#if USE_TDM
// 2D TDM load: tile_dim0=16 (contiguous), tile_dim1=1024 rows, row stride 32
// elements, data_size=4B.  D# per CDNA5 ISA.
// This toolchain (clang-23 / therock) exposes the 6-arg builtin:
//   (u32x4 g0, i32x8 g1, i32x4 g2, i32x4 g3, i32x8 extra, i32 cpol)
static __device__ __forceinline__ void tdm_load_pre(const float* gp, unsigned lds_off) {
  unsigned long long ga = (unsigned long long)(size_t)gp;
  u32x4 g0;
  g0[0] = 1u;                                   // count=1, user descriptor
  g0[1] = lds_off;                              // lds_addr
  g0[2] = (unsigned)ga;                         // global_addr[31:0]
  g0[3] = (unsigned)((ga >> 32) & 0x01FFFFFFull) | 0x80000000u;  // addr[56:32] | type=2
  i32x8 g1;
  g1[0] = 0x00020000;                           // data_size=4B, no flags
  g1[1] = (16 << 16);                           // tensor_dim0 = 16
  g1[2] = (1024 << 16);                         // tensor_dim1 = 1024
  g1[3] = (16 << 16);                           // tile_dim0 = 16
  g1[4] = 1024;                                 // tile_dim1 = 1024
  g1[5] = 32;                                   // tensor_dim0_stride = 32
  g1[6] = 0; g1[7] = 0;
  i32x4 gz; gz[0] = 0; gz[1] = 0; gz[2] = 0; gz[3] = 0;
  i32x8 gz8;
#pragma unroll
  for (int j = 0; j < 8; ++j) gz8[j] = 0;
  __builtin_amdgcn_tensor_load_to_lds(g0, g1, gz, gz, gz8, 0);
}
#endif

// ---------------------------------------------------------------------------
// Kernel 1: weight prep (bf16 hi/lo; conv_w re-laid [d][kt][c]; fused bias).
// ---------------------------------------------------------------------------
__global__ __launch_bounds__(256) void etb_prep(
    const float* __restrict__ w_ih, const float* __restrict__ w_hh,
    const float* __restrict__ conv_w, const float* __restrict__ b_ih,
    const float* __restrict__ b_hh,
    void* wih_hi_, void* wih_lo_, void* whh_hi_, void* whh_lo_,
    void* wc_hi_, void* wc_lo_, float* __restrict__ bvec) {
  __bf16* wih_hi = (__bf16*)wih_hi_; __bf16* wih_lo = (__bf16*)wih_lo_;
  __bf16* whh_hi = (__bf16*)whh_hi_; __bf16* whh_lo = (__bf16*)whh_lo_;
  __bf16* wc_hi  = (__bf16*)wc_hi_;  __bf16* wc_lo  = (__bf16*)wc_lo_;
  int idx = blockIdx.x * 256 + threadIdx.x;
  if (idx < G4 * Dsz) {
    float v = w_ih[idx]; __bf16 h = (__bf16)v;
    wih_hi[idx] = h; wih_lo[idx] = (__bf16)(v - (float)h);
  }
  if (idx < G4 * Hsz) {
    float v = w_hh[idx]; __bf16 h = (__bf16)v;
    whh_hi[idx] = h; whh_lo[idx] = (__bf16)(v - (float)h);
  }
  if (idx < Dsz * KC) {
    int d = idx / KC, r = idx % KC;
    int kt = r >> 7, c = r & 127;
    float v = conv_w[(size_t)d * KC + c * 3 + kt];
    __bf16 h = (__bf16)v;
    wc_hi[idx] = h; wc_lo[idx] = (__bf16)(v - (float)h);
  }
  if (idx < G4) bvec[idx] = b_ih[idx] + b_hh[idx];
}

// ---------------------------------------------------------------------------
// Kernel 2: x -> xhi/xlo bf16 split (done ONCE; removes all split VALU from
// the GEMM / conv inner loops, which re-read x 16x / 24x).
// ---------------------------------------------------------------------------
__global__ __launch_bounds__(256) void etb_split_x(
    const float* __restrict__ x, void* xhi_, void* xlo_) {
  __bf16* xhi = (__bf16*)xhi_; __bf16* xlo = (__bf16*)xlo_;
  size_t idx = (size_t)blockIdx.x * 256 + threadIdx.x;   // B*T*D = 8388608
  float v = x[idx];
  __bf16 h = (__bf16)v;
  xhi[idx] = h; xlo[idx] = (__bf16)(v - (float)h);
}

// ---------------------------------------------------------------------------
// Kernel 3: pre = x @ w_ih^T + bias.  M=16384, N=1024, K=512.
// Wave computes 16x64 (4 n-tiles): A-frag reused 4x. Output in [t][n][b].
// ---------------------------------------------------------------------------
__global__ __launch_bounds__(256) void etb_gemm_pre(
    const void* xhi_, const void* xlo_, const void* wih_hi_, const void* wih_lo_,
    const float* __restrict__ bvec, float* __restrict__ pre) {
  const __bf16* xhi = (const __bf16*)xhi_;
  const __bf16* xlo = (const __bf16*)xlo_;
  const __bf16* wih_hi = (const __bf16*)wih_hi_;
  const __bf16* wih_lo = (const __bf16*)wih_lo_;
  const int lane = threadIdx.x & 31, wave = threadIdx.x >> 5;
  const int tile = blockIdx.x * 8 + wave;            // 16384 wave-tiles
  const int mt = tile >> 4, ng = tile & 15;
  const int mbase = mt * 16, nbase = ng * 64;
  const int l15 = lane & 15, khalf = lane >> 4;
  const size_t arow = (size_t)(mbase + l15) * Dsz;

  v8f acc[4];
#pragma unroll
  for (int u = 0; u < 4; ++u) {
    float bias = bvec[nbase + u * 16 + l15];
#pragma unroll
    for (int i = 0; i < 8; ++i) acc[u][i] = bias;
  }

  for (int kb = 0; kb < Dsz; kb += 32) {
    const __bf16* ah = xhi + arow + kb + khalf * 8;
    const __bf16* al = xlo + arow + kb + khalf * 8;
    __builtin_prefetch(ah + 32, 0, 0);               // global_prefetch_b8
    v16bf ahi = afrag(ah), alo = afrag(al);
#pragma unroll
    for (int u = 0; u < 4; ++u) {
      const size_t brow = (size_t)(nbase + u * 16 + l15) * Dsz + kb + khalf * 8;
      v16bf bhi = bfrag(wih_hi + brow);
      v16bf blo = bfrag(wih_lo + brow);
      acc[u] = wmma_bf16(ahi, bhi, acc[u]);
      acc[u] = wmma_bf16(ahi, blo, acc[u]);
      acc[u] = wmma_bf16(alo, bhi, acc[u]);
    }
  }

#pragma unroll
  for (int u = 0; u < 4; ++u) {
    const int n = nbase + u * 16 + l15;
#pragma unroll
    for (int i = 0; i < 8; ++i) {
      int m = mbase + khalf * 8 + i;                 // C layout: m=(lane/16)*8+i
      int b = m >> 9, t = m & 511;                   // m = b*T + t
      pre[((size_t)t * G4 + n) * Bsz + b] = acc[u][i];
    }
  }
}

// ---------------------------------------------------------------------------
// Kernel 4: grouped dilated conv (implicit GEMM) + bias + exact GELU.
// Wave computes 16 channels x 64 timesteps: weight A-frag reused 4x.
// ---------------------------------------------------------------------------
__global__ __launch_bounds__(256) void etb_conv_gelu(
    const void* xhi_, const void* xlo_, const void* wc_hi_, const void* wc_lo_,
    const float* __restrict__ conv_b, float* __restrict__ y) {
  const __bf16* xhi = (const __bf16*)xhi_;
  const __bf16* xlo = (const __bf16*)xlo_;
  const __bf16* wc_hi = (const __bf16*)wc_hi_;
  const __bf16* wc_lo = (const __bf16*)wc_lo_;
  const int lane = threadIdx.x & 31, wave = threadIdx.x >> 5;
  const int tile = blockIdx.x * 8 + wave;            // 8192 wave-tiles
  const int b = tile >> 8, dt = (tile >> 3) & 31, tg = tile & 7;
  const int dbase = dt * 16, tg64 = tg * 64;
  const int l15 = lane & 15, khalf = lane >> 4;
  const int g = dbase >> 7;

  v8f acc[4];
#pragma unroll
  for (int u = 0; u < 4; ++u)
#pragma unroll
    for (int i = 0; i < 8; ++i) acc[u][i] = conv_b[dbase + khalf * 8 + i];

  const size_t wrow = (size_t)(dbase + l15) * KC;
  for (int kb = 0; kb < KC; kb += 32) {
    v16bf ahi = afrag(wc_hi + wrow + kb + khalf * 8);
    v16bf alo = afrag(wc_lo + wrow + kb + khalf * 8);
    const int kt = kb >> 7;                          // tap 0..2 -> t offset -2/0/+2
    const size_t coff = (size_t)g * 128 + (kb & 127) + khalf * 8;
#pragma unroll
    for (int u = 0; u < 4; ++u) {
      const int trow = tg64 + u * 16 + l15 + 2 * (kt - 1);
      v16bf bhi, blo;
#pragma unroll
      for (int j = 0; j < 16; ++j) { bhi[j] = (__bf16)0.f; blo[j] = (__bf16)0.f; }
      if (trow >= 0 && trow < Tsz) {
        const size_t boff = ((size_t)b * Tsz + trow) * Dsz + coff;
        bhi = bfrag(xhi + boff);
        blo = bfrag(xlo + boff);
      }
      acc[u] = wmma_bf16(ahi, bhi, acc[u]);
      acc[u] = wmma_bf16(ahi, blo, acc[u]);
      acc[u] = wmma_bf16(alo, bhi, acc[u]);
    }
  }

#pragma unroll
  for (int u = 0; u < 4; ++u) {
    const int t = tg64 + u * 16 + l15;
#pragma unroll
    for (int i = 0; i < 8; ++i) {
      int d = dbase + khalf * 8 + i;
      float v = acc[u][i];
      v = 0.5f * v * (1.f + erff(v * 0.70710678118654752f));
      y[((size_t)b * Dsz + d) * Tsz + t] = v;
    }
  }
}

// ---------------------------------------------------------------------------
// Kernel 5: row norm over T per (b,d), unbiased std, gamma/beta indexed by t
// (reference's T==D broadcast bug), transposed store into d_out.
// ---------------------------------------------------------------------------
__global__ __launch_bounds__(256) void etb_norm(
    const float* __restrict__ y, const float* __restrict__ gamma,
    const float* __restrict__ beta, float* __restrict__ out) {
  __shared__ float s1[256], s2[256];
  const int bd = blockIdx.x;
  const int b = bd >> 9, d = bd & 511;
  const float* row = y + (size_t)bd * Tsz;
  const int tid = threadIdx.x;
  float a0 = row[tid], a1 = row[tid + 256];
  s1[tid] = a0 + a1;
  s2[tid] = a0 * a0 + a1 * a1;
  __syncthreads();
  for (int s = 128; s > 0; s >>= 1) {
    if (tid < s) { s1[tid] += s1[tid + s]; s2[tid] += s2[tid + s]; }
    __syncthreads();
  }
  float mean = s1[0] * (1.f / 512.f);
  float var  = fmaxf((s2[0] - 512.f * mean * mean) * (1.f / 511.f), 0.f);
  float inv  = 1.f / (sqrtf(var) + 1e-6f);
  out[((size_t)b * Tsz + tid) * Dsz + d]       = gamma[tid] * (a0 - mean) * inv + beta[tid];
  out[((size_t)b * Tsz + tid + 256) * Dsz + d] = gamma[tid + 256] * (a1 - mean) * inv + beta[tid + 256];
}

// ---------------------------------------------------------------------------
// Kernel 6: persistent LSTM scan. 2 blocks x 512 threads (16 waves), block
// owns 16 batch rows. Wave w owns gate tiles {w,w+16,w+32,w+48} so i,f,g,o for
// the same h-columns land in its C-frags; c stays in registers for all 512
// steps; h double-buffered in LDS (bf16 hi/lo). TDM prefetches next step's
// pre[t+1][:, bbase:bbase+16] slice (64KB) into double-buffered dynamic LDS,
// overlapped with the step's 96 WMMAs; completion via s_wait_tensorcnt.
// ---------------------------------------------------------------------------
__global__ __launch_bounds__(512) void etb_lstm(
    const float* __restrict__ pre, const void* whh_hi_, const void* whh_lo_,
    float* __restrict__ out) {
  const __bf16* whh_hi = (const __bf16*)whh_hi_;
  const __bf16* whh_lo = (const __bf16*)whh_lo_;
  __shared__ __align__(16) __bf16 sh_hi[2 * 16 * 256];
  __shared__ __align__(16) __bf16 sh_lo[2 * 16 * 256];
  extern __shared__ __align__(16) float pre_buf[];   // 2 x 16384 floats (128KB)

  const int lane = threadIdx.x & 31, wave = threadIdx.x >> 5;  // wave 0..15
  const int bbase = blockIdx.x * 16;
  for (int i = threadIdx.x; i < 2 * 16 * 256; i += 512) {
    sh_hi[i] = (__bf16)0.f; sh_lo[i] = (__bf16)0.f;            // h0 = 0
  }
  float cst[8];
#pragma unroll
  for (int i = 0; i < 8; ++i) cst[i] = 0.f;                    // c0 = 0 (regs)

  const int l15 = lane & 15, khalf = lane >> 4;
  const int jcol = wave * 16 + l15;                            // h column
  const int mloc = khalf * 8;                                  // C-frag row base

#if USE_TDM
  const unsigned lds_base = (unsigned)(size_t)(void*)pre_buf;
  if (wave == 0) {                                             // prologue: load t=0
    tdm_load_pre(pre + (size_t)0 * G4 * Bsz + bbase, lds_base);
    __builtin_amdgcn_s_wait_tensorcnt(0);
  }
#endif
  __syncthreads();

  for (int t = 0; t < Tsz; ++t) {
    const int cur = t & 1, nxt = cur ^ 1;
#if USE_TDM
    if (wave == 0 && t + 1 < Tsz)                              // prefetch t+1
      tdm_load_pre(pre + (size_t)(t + 1) * G4 * Bsz + bbase,
                   lds_base + (unsigned)nxt * 65536u);
#endif
    v8f gq[4];
#pragma unroll
    for (int q = 0; q < 4; ++q) {                              // gate init (i,f,g,o)
#if USE_TDM
      const float* pg = pre_buf + cur * 16384 + (q * 256 + jcol) * 16 + mloc;
#else
      const float* pg = pre + ((size_t)t * G4 + q * 256 + jcol) * Bsz + bbase + mloc;
#endif
      float4 p0 = *(const float4*)pg;
      float4 p1 = *(const float4*)(pg + 4);
      gq[q][0] = p0.x; gq[q][1] = p0.y; gq[q][2] = p0.z; gq[q][3] = p0.w;
      gq[q][4] = p1.x; gq[q][5] = p1.y; gq[q][6] = p1.z; gq[q][7] = p1.w;
    }
    for (int kb = 0; kb < Hsz; kb += 32) {
      const int aoff = cur * 4096 + l15 * 256 + kb + khalf * 8;
      v16bf ahi = cat8(*(const v8bf*)&sh_hi[aoff], *(const v8bf*)&sh_hi[aoff + 16]);
      v16bf alo = cat8(*(const v8bf*)&sh_lo[aoff], *(const v8bf*)&sh_lo[aoff + 16]);
#pragma unroll
      for (int q = 0; q < 4; ++q) {
        const size_t brow = (size_t)(q * 256 + jcol) * Hsz + kb + khalf * 8;
        v16bf bhi = bfrag(whh_hi + brow);
        v16bf blo = bfrag(whh_lo + brow);
        gq[q] = wmma_bf16(ahi, bhi, gq[q]);
        gq[q] = wmma_bf16(ahi, blo, gq[q]);
        gq[q] = wmma_bf16(alo, bhi, gq[q]);
      }
    }
#pragma unroll
    for (int i = 0; i < 8; ++i) {                              // pointwise in regs
      float iv = 1.f / (1.f + expf(-gq[0][i]));
      float fv = 1.f / (1.f + expf(-gq[1][i]));
      float gv = tanhf(gq[2][i]);
      float ov = 1.f / (1.f + expf(-gq[3][i]));
      cst[i] = fv * cst[i] + iv * gv;
      float h = ov * tanhf(cst[i]);
      int m = mloc + i;
      out[(((size_t)(bbase + m)) * Tsz + t) * Dsz + jcol] += h;
      __bf16 hh = (__bf16)h;
      sh_hi[nxt * 4096 + m * 256 + jcol] = hh;
      sh_lo[nxt * 4096 + m * 256 + jcol] = (__bf16)(h - (float)hh);
    }
#if USE_TDM
    if (wave == 0 && t + 1 < Tsz)
      __builtin_amdgcn_s_wait_tensorcnt(0);                    // t+1 slice landed
#endif
    __syncthreads();                                           // one barrier/step
  }
}

// ---------------------------------------------------------------------------
extern "C" void kernel_launch(void* const* d_in, const int* in_sizes, int n_in,
                              void* d_out, int out_size, void* d_ws, size_t ws_size,
                              hipStream_t stream) {
  const float* x      = (const float*)d_in[0];
  const float* conv_w = (const float*)d_in[1];
  const float* conv_b = (const float*)d_in[2];
  const float* gamma  = (const float*)d_in[3];
  const float* beta   = (const float*)d_in[4];
  const float* w_ih   = (const float*)d_in[5];
  const float* w_hh   = (const float*)d_in[6];
  const float* b_ih   = (const float*)d_in[7];
  const float* b_hh   = (const float*)d_in[8];
  float* out = (float*)d_out;

  char* ws = (char*)d_ws;
  size_t off = 0;
  float* y   = (float*)(ws + off); off += (size_t)Bsz * Dsz * Tsz * 4;   // 32 MB
  float* pre = (float*)(ws + off); off += (size_t)Tsz * G4 * Bsz * 4;    // 64 MB
  void* xhi = ws + off; off += (size_t)Bsz * Tsz * Dsz * 2;              // 16 MB
  void* xlo = ws + off; off += (size_t)Bsz * Tsz * Dsz * 2;              // 16 MB
  void* wih_hi = ws + off; off += (size_t)G4 * Dsz * 2;
  void* wih_lo = ws + off; off += (size_t)G4 * Dsz * 2;
  void* whh_hi = ws + off; off += (size_t)G4 * Hsz * 2;
  void* whh_lo = ws + off; off += (size_t)G4 * Hsz * 2;
  void* wc_hi  = ws + off; off += (size_t)Dsz * KC * 2;
  void* wc_lo  = ws + off; off += (size_t)Dsz * KC * 2;
  float* bvec  = (float*)(ws + off); off += (size_t)G4 * 4;

  etb_prep<<<2048, 256, 0, stream>>>(w_ih, w_hh, conv_w, b_ih, b_hh,
                                     wih_hi, wih_lo, whh_hi, whh_lo,
                                     wc_hi, wc_lo, bvec);
  etb_split_x<<<32768, 256, 0, stream>>>(x, xhi, xlo);
  etb_gemm_pre<<<2048, 256, 0, stream>>>(xhi, xlo, wih_hi, wih_lo, bvec, pre);
  etb_conv_gelu<<<1024, 256, 0, stream>>>(xhi, xlo, wc_hi, wc_lo, conv_b, y);
  etb_norm<<<16384, 256, 0, stream>>>(y, gamma, beta, out);
  etb_lstm<<<2, 512, 2 * 16384 * sizeof(float), stream>>>(pre, whh_hi, whh_lo, out);
}
